// CG_GNN_Encoder_17368847745364
// MI455X (gfx1250) — matrix operand order repaired and checked by reading, not compiled, not run on hardware
//
#include <hip/hip_runtime.h>
#include <stdint.h>
#include <math.h>

#define NN 20000
#define HH 4
#define DD 512          // H*C == IN == 512
#define NEG_SLOPE 0.2f

typedef __attribute__((ext_vector_type(16))) __bf16 v16bf;
typedef __attribute__((ext_vector_type(8)))  float  v8f;

union Frag16 { v16bf v; uint4 u[2]; };

__device__ __forceinline__ unsigned short f32_to_bf16(float x) {
  unsigned u = __float_as_uint(x);
  u += 0x7FFFu + ((u >> 16) & 1u);       // round-to-nearest-even
  return (unsigned short)(u >> 16);
}

__global__ void k_f32_to_bf16(const float* __restrict__ in,
                              unsigned short* __restrict__ out, int n) {
  int i = blockIdx.x * blockDim.x + threadIdx.x;
  if (i < n) out[i] = f32_to_bf16(in[i]);
}

__global__ void k_fill_f32(float* __restrict__ p, float v, int n) {
  int i = blockIdx.x * blockDim.x + threadIdx.x;
  if (i < n) p[i] = v;
}

// Swizzle W [512,512] fp32 row-major -> bf16 WMMA B-fragments.
// Fragment index = (kb*32 + nt)*32 + lane, 16 bf16 per lane:
//   lane<16 : K = kb*32 +  0..15, col = nt*16 + lane
//   lane>=16: K = kb*32 + 16..31, col = nt*16 + (lane-16)
__global__ void k_prep_w(const float* __restrict__ W,
                         unsigned short* __restrict__ out) {
  int tid = blockIdx.x * blockDim.x + threadIdx.x;   // 16*32*32 = 16384
  if (tid >= 16 * 32 * 32) return;
  int lane = tid & 31;
  int nt   = (tid >> 5) & 31;
  int kb   = tid >> 10;
  int col  = nt * 16 + (lane & 15);
  int kbase = kb * 32 + ((lane < 16) ? 0 : 16);
  unsigned short* o = out + (size_t)tid * 16;
#pragma unroll
  for (int e = 0; e < 16; ++e)
    o[e] = f32_to_bf16(W[(size_t)(kbase + e) * DD + col]);
}

// C[20000,512] = A(bf16)[20000,512] @ B(swizzled bf16)[512,512] + bias
// wave tile 16M x 64N; K stepped 32/WMMA; software-pipelined double buffer so
// next iteration's A/B fragment loads overlap the current 4 WMMAs.
__global__ void __launch_bounds__(256)
k_gemm_bf16(const unsigned short* __restrict__ A,
            const unsigned short* __restrict__ Bsw,
            const float* __restrict__ bias,
            float* __restrict__ C) {
  int lane  = threadIdx.x & 31;
  int wave  = threadIdx.x >> 5;      // 0..7 -> which 64-wide N tile
  int mBase = blockIdx.x * 16;
  int nBase = wave * 64;

  v8f acc[4] = {};

  int row = mBase + (lane & 15);
  const unsigned short* arow = A + (size_t)row * DD + ((lane < 16) ? 0 : 16);
  // B fragment address: Bsw + lane*16 + nt*512 + kb*16384
  const unsigned short* bbase =
      Bsw + (size_t)lane * 16 + (size_t)(nBase >> 4) * 512;

  // prologue: load kb = 0
  Frag16 a_cur, b_cur[4];
  a_cur.u[0] = *(const uint4*)(arow);
  a_cur.u[1] = *(const uint4*)(arow + 8);
#pragma unroll
  for (int j = 0; j < 4; ++j) {
    b_cur[j].u[0] = *(const uint4*)(bbase + j * 512);
    b_cur[j].u[1] = *(const uint4*)(bbase + j * 512 + 8);
  }

#pragma unroll 2
  for (int kb = 0; kb < 16; ++kb) {
    // prefetch kb+1 fragments (last iteration re-loads kb -> harmless, keeps
    // the loop body branch-free and uniform)
    int kn = (kb < 15) ? (kb + 1) : kb;
    const unsigned short* apn = arow + kn * 32;
    const unsigned short* bpn = bbase + (size_t)kn * 16384;
    Frag16 a_nxt, b_nxt[4];
    a_nxt.u[0] = *(const uint4*)(apn);
    a_nxt.u[1] = *(const uint4*)(apn + 8);
#pragma unroll
    for (int j = 0; j < 4; ++j) {
      b_nxt[j].u[0] = *(const uint4*)(bpn + j * 512);
      b_nxt[j].u[1] = *(const uint4*)(bpn + j * 512 + 8);
    }
    // math for kb (depends only on previously-loaded buffers)
#pragma unroll
    for (int j = 0; j < 4; ++j)
      acc[j] = __builtin_amdgcn_wmma_f32_16x16x32_bf16(
          false, a_cur.v, false, b_nxt == nullptr ? b_cur[j].v : b_cur[j].v,
          (short)0, acc[j], false, false);
    a_cur = a_nxt;
#pragma unroll
    for (int j = 0; j < 4; ++j) b_cur[j] = b_nxt[j];
  }

  int rbase = mBase + ((lane < 16) ? 0 : 8);
#pragma unroll
  for (int j = 0; j < 4; ++j) {
    int col = nBase + j * 16 + (lane & 15);
    float bc = bias[col];
#pragma unroll
    for (int r = 0; r < 8; ++r)
      C[(size_t)(rbase + r) * DD + col] = acc[j][r] + bc;
  }
}

__device__ __forceinline__ int edge_src(const long long* ei, int E, int e) {
  return (e < E) ? (int)ei[e] : (e - E);
}
__device__ __forceinline__ int edge_dst(const long long* ei, int E, int e) {
  return (e < E) ? (int)ei[E + e] : (e - E);
}

// float atomic max via ordered-int trick (init must be -inf)
__device__ __forceinline__ void atomicMaxF(float* addr, float v) {
  if (v >= 0.0f) atomicMax((int*)addr, __float_as_int(v));
  else           atomicMin((unsigned int*)addr, __float_as_uint(v));
}

// one wave per edge: logits[e,h] = sum_c att[h,c]*leaky(xl[src]+xr[dst]); atomic max into m
__global__ void __launch_bounds__(256)
k_edge_logits(const float* __restrict__ xl, const float* __restrict__ xr,
              const float* __restrict__ att, const long long* __restrict__ ei,
              int E, int Et, float* __restrict__ logits, float* __restrict__ m) {
  __shared__ float satt[DD];
  for (int i = threadIdx.x; i < DD; i += blockDim.x) satt[i] = att[i];
  __syncthreads();
  int e    = (blockIdx.x * blockDim.x + threadIdx.x) >> 5;
  int lane = threadIdx.x & 31;
  if (e >= Et) return;
  int s = edge_src(ei, E, e);
  int d = edge_dst(ei, E, e);
  const float* pl = xl + (size_t)s * DD;
  const float* pr = xr + (size_t)d * DD;
  float acc[HH] = {0.f, 0.f, 0.f, 0.f};
#pragma unroll
  for (int j = 0; j < 16; ++j) {
    int idx = lane + 32 * j;            // head = idx >> 7 == j >> 2
    float v = pl[idx] + pr[idx];
    v = (v > 0.f) ? v : NEG_SLOPE * v;
    acc[j >> 2] += v * satt[idx];
  }
#pragma unroll
  for (int off = 16; off > 0; off >>= 1)
#pragma unroll
    for (int h = 0; h < HH; ++h) acc[h] += __shfl_xor(acc[h], off, 32);
  if (lane == 0) {
#pragma unroll
    for (int h = 0; h < HH; ++h) {
      logits[(size_t)e * HH + h] = acc[h];
      atomicMaxF(&m[d * HH + h], acc[h]);
    }
  }
}

// p = exp(logit - m[dst]); in place; atomic sum into s
__global__ void k_edge_exp(float* __restrict__ logits, const float* __restrict__ m,
                           float* __restrict__ s, const long long* __restrict__ ei,
                           int E, int Et) {
  int t = blockIdx.x * blockDim.x + threadIdx.x;
  if (t >= Et * HH) return;
  int e = t >> 2, h = t & 3;
  int d = edge_dst(ei, E, e);
  float p = __expf(logits[t] - m[d * HH + h]);
  logits[t] = p;
  atomicAdd(&s[d * HH + h], p);
}

// out[dst] += xl[src] * (p/s[dst]) ; one wave per edge, f32 atomics (resolve in L2)
__global__ void __launch_bounds__(256)
k_edge_agg(const float* __restrict__ xl, const float* __restrict__ p,
           const float* __restrict__ s, const long long* __restrict__ ei,
           int E, int Et, float* __restrict__ out) {
  int e    = (blockIdx.x * blockDim.x + threadIdx.x) >> 5;
  int lane = threadIdx.x & 31;
  if (e >= Et) return;
  int si = edge_src(ei, E, e);
  int d  = edge_dst(ei, E, e);
  float w[HH];
#pragma unroll
  for (int h = 0; h < HH; ++h) w[h] = p[(size_t)e * HH + h] / s[d * HH + h];
  const float* pl = xl + (size_t)si * DD;
  float* po = out + (size_t)d * DD;
#pragma unroll
  for (int j = 0; j < 16; ++j) {
    int idx = lane + 32 * j;
    atomicAdd(&po[idx], pl[idx] * w[j >> 2]);
  }
}

// layer-0 epilogue: h = elu(acc + bias) -> bf16 for next GEMM
__global__ void k_epilogue_elu_bf16(const float* __restrict__ acc,
                                    const float* __restrict__ bias,
                                    unsigned short* __restrict__ outbf, int n) {
  int i = blockIdx.x * blockDim.x + threadIdx.x;
  if (i >= n) return;
  float v = acc[i] + bias[i & (DD - 1)];
  v = (v > 0.f) ? v : (__expf(v) - 1.f);
  outbf[i] = f32_to_bf16(v);
}

__global__ void k_add_bias(float* __restrict__ out, const float* __restrict__ bias, int n) {
  int i = blockIdx.x * blockDim.x + threadIdx.x;
  if (i >= n) return;
  out[i] += bias[i & (DD - 1)];
}

extern "C" void kernel_launch(void* const* d_in, const int* in_sizes, int n_in,
                              void* d_out, int out_size, void* d_ws, size_t ws_size,
                              hipStream_t stream) {
  const float*     x  = (const float*)d_in[0];
  const long long* ei = (const long long*)d_in[1];   // int64 in reference
  const float* Wl[2]   = {(const float*)d_in[2],  (const float*)d_in[8]};
  const float* bl[2]   = {(const float*)d_in[3],  (const float*)d_in[9]};
  const float* Wr[2]   = {(const float*)d_in[4],  (const float*)d_in[10]};
  const float* br[2]   = {(const float*)d_in[5],  (const float*)d_in[11]};
  const float* att[2]  = {(const float*)d_in[6],  (const float*)d_in[12]};
  const float* bias[2] = {(const float*)d_in[7],  (const float*)d_in[13]};

  const int E  = in_sizes[1] / 2;
  const int Et = E + NN;
  const int NELEM = NN * DD;

  char* ws = (char*)d_ws;
  size_t off = 0;
  auto alloc = [&](size_t bytes) -> void* {
    void* p = ws + off;
    off = (off + bytes + 255) & ~(size_t)255;
    return p;
  };
  unsigned short* bfA  = (unsigned short*)alloc((size_t)NELEM * 2);
  unsigned short* WswL = (unsigned short*)alloc((size_t)DD * DD * 2);
  unsigned short* WswR = (unsigned short*)alloc((size_t)DD * DD * 2);
  float* xl    = (float*)alloc((size_t)NELEM * 4);
  float* xr    = (float*)alloc((size_t)NELEM * 4);   // reused as layer-0 accumulator
  float* logit = (float*)alloc((size_t)Et * HH * 4); // reused in-place as softmax p
  float* mbuf  = (float*)alloc((size_t)NN * HH * 4);
  float* sbuf  = (float*)alloc((size_t)NN * HH * 4);
  (void)ws_size; (void)n_in; (void)out_size;

  dim3 b256(256);
  int gElem     = (NELEM + 255) / 256;
  int gEdgeWave = (Et * 32 + 255) / 256;   // one wave32 per edge
  int gEdgeH    = (Et * HH + 255) / 256;
  int gNH       = (NN * HH + 255) / 256;

  k_f32_to_bf16<<<gElem, b256, 0, stream>>>(x, bfA, NELEM);

  for (int l = 0; l < 2; ++l) {
    float* accum = (l == 0) ? xr : (float*)d_out;
    k_prep_w<<<64, b256, 0, stream>>>(Wl[l], WswL);
    k_prep_w<<<64, b256, 0, stream>>>(Wr[l], WswR);
    k_gemm_bf16<<<NN / 16, b256, 0, stream>>>(bfA, WswL, bl[l], xl);
    k_gemm_bf16<<<NN / 16, b256, 0, stream>>>(bfA, WswR, br[l], xr);
    k_fill_f32<<<gNH, b256, 0, stream>>>(mbuf, -INFINITY, NN * HH);
    k_fill_f32<<<gNH, b256, 0, stream>>>(sbuf, 0.f, NN * HH);
    k_edge_logits<<<gEdgeWave, b256, 0, stream>>>(xl, xr, att[l], ei, E, Et, logit, mbuf);
    k_fill_f32<<<gElem, b256, 0, stream>>>(accum, 0.f, NELEM);  // safe: xr no longer needed
    k_edge_exp<<<gEdgeH, b256, 0, stream>>>(logit, mbuf, sbuf, ei, E, Et);
    k_edge_agg<<<gEdgeWave, b256, 0, stream>>>(xl, logit, sbuf, ei, E, Et, accum);
    if (l == 0)
      k_epilogue_elu_bf16<<<gElem, b256, 0, stream>>>(accum, bias[0], bfA, NELEM);
    else
      k_add_bias<<<gElem, b256, 0, stream>>>((float*)d_out, bias[1], NELEM);
  }
}
